// Attention_19868518711592
// MI455X (gfx1250) — matrix operand order repaired and checked by reading, not compiled, not run on hardware
//
#include <hip/hip_runtime.h>

typedef __attribute__((ext_vector_type(16))) _Float16 v16h;
typedef __attribute__((ext_vector_type(8)))  _Float16 v8h;
typedef __attribute__((ext_vector_type(2)))  _Float16 v2h;
typedef __attribute__((ext_vector_type(8)))  float    v8f;
typedef __attribute__((ext_vector_type(4)))  float    v4f;

#define ATT_B 4
#define ATT_H 16
#define ATT_S 2048
#define ATT_D 128
#define BQ 128          // q rows per workgroup
#define BK 64           // keys per tile
#define NWAVES 8

__device__ __forceinline__ v16h cat8(v8h a, v8h b) {
    return __builtin_shufflevector(a, b, 0,1,2,3,4,5,6,7,8,9,10,11,12,13,14,15);
}

__device__ __forceinline__ v8f wmma_f16(v16h a, v16h b, v8f c) {
    return __builtin_amdgcn_wmma_f32_16x16x32_f16(false, a, false, b,
                                                  (short)0, c, false, false);
}

// One 64-key tile of flash attention for one wave (16 q rows).
// Scores arrive pre-scaled by (1/sqrt(D))*log2(e): softmax is pure exp2.
// Row-sums are accumulated by an extra P*ones WMMA (accL) instead of
// cross-lane shuffles; only the row-max still needs a butterfly.
template<bool MASK>
__device__ __forceinline__ void attn_tile(
    const _Float16* __restrict__ sK, const _Float16* __restrict__ sVt,
    _Float16* __restrict__ pp,
    const v16h (&qa)[4], v8f (&acc)[8], v8f& accL, float (&mrun)[8],
    int kb, int qrow0, int l16, int h16)
{
    // ---- S = Q K^T : four 16x16 score frags ----
    v8f sfrag[4];
#pragma unroll
    for (int st = 0; st < 4; ++st) {
        v8f s = (v8f)0.0f;
#pragma unroll
        for (int c = 0; c < 4; ++c) {
            // B layout: lane (h,n=l16) holds Kmat[key st*16+n][d = c*32+h*8 .. +7, c*32+16+h*8 .. +7]
            const _Float16* bp = &sK[(st * 16 + l16) * ATT_D + c * 32 + h16 * 8];
            v16h bb = cat8(*(const v8h*)bp, *(const v8h*)(bp + 16));
            s = wmma_f16(qa[c], bb, s);
        }
        sfrag[st] = s;
    }

    // ---- online softmax (log2 domain), row-max only ----
#pragma unroll
    for (int v = 0; v < 8; ++v) {
        float a[4];
#pragma unroll
        for (int st = 0; st < 4; ++st) a[st] = sfrag[st][v];
        if (MASK) {
            const int qm = qrow0 + v + 8 * h16;
#pragma unroll
            for (int st = 0; st < 4; ++st)
                if (kb + st * 16 + l16 > qm) a[st] = -1e30f;
        }
        float mx = fmaxf(fmaxf(a[0], a[1]), fmaxf(a[2], a[3]));
#pragma unroll
        for (int off = 1; off < 16; off <<= 1)
            mx = fmaxf(mx, __shfl_xor(mx, off, 32));
        const float mnew = fmaxf(mrun[v], mx);
        const float corr = exp2f(mrun[v] - mnew);
        mrun[v] = mnew;
#pragma unroll
        for (int f = 0; f < 8; ++f) acc[f][v] *= corr;
        accL[v] *= corr;
        // stash P in C-order (wave-private LDS; same-wave DS ops are in-order)
#pragma unroll
        for (int st = 0; st < 4; ++st)
            pp[(v + 8 * h16) * BK + st * 16 + l16] = (_Float16)exp2f(a[st] - mnew);
    }

    // ---- reload P as two A-layout frags (keys 0-31, 32-63) ----
    const _Float16* pr = pp + l16 * BK;
    v16h pa0 = cat8(*(const v8h*)(pr + h16 * 8),      *(const v8h*)(pr + 16 + h16 * 8));
    v16h pa1 = cat8(*(const v8h*)(pr + 32 + h16 * 8), *(const v8h*)(pr + 48 + h16 * 8));

    // ---- O += P V ----
#pragma unroll
    for (int f = 0; f < 8; ++f) {
        const _Float16* vb = &sVt[(f * 16 + l16) * BK];
        v16h b0 = cat8(*(const v8h*)(vb + h16 * 8),
                       *(const v8h*)(vb + 16 + h16 * 8));
        v16h b1 = cat8(*(const v8h*)(vb + 32 + h16 * 8),
                       *(const v8h*)(vb + 48 + h16 * 8));
        acc[f] = wmma_f16(pa0, b0, acc[f]);
        acc[f] = wmma_f16(pa1, b1, acc[f]);
    }

    // ---- row-sum via ones-matrix WMMA: accL += P x 1 (replicated per lane) ----
    v16h ones;
#pragma unroll
    for (int j = 0; j < 16; ++j) ones[j] = (_Float16)1.0f;
    accL = wmma_f16(pa0, ones, accL);
    accL = wmma_f16(pa1, ones, accL);
}

__global__ __launch_bounds__(256, 1)
void fa2_wmma_kernel(const float* __restrict__ Qg,
                     const float* __restrict__ Kg,
                     const float* __restrict__ Vg,
                     float* __restrict__ Og)
{
    // single-buffer 64-key staging + per-wave P scratch: 48 KB
    __shared__ __align__(16) _Float16 sK [BK * ATT_D];        // [key][d]
    __shared__ __align__(16) _Float16 sVt[ATT_D * BK];        // [d][key]
    __shared__ __align__(16) _Float16 sP [NWAVES * 16 * BK];  // per-wave P

    const int tid  = threadIdx.x;
    const int lane = tid & 31;
    const int wave = tid >> 5;
    const int h16  = lane >> 4;
    const int l16  = lane & 15;

    const int qb = blockIdx.x;    // q tile 0..15
    const int bh = blockIdx.y;    // fused b*H + h, 0..63

    const size_t base = (size_t)bh * ATT_S * ATT_D;
    const float* Qp = Qg + base;
    const float* Kp = Kg + base;
    const float* Vp = Vg + base;
    float*       Op = Og + base;

    const int qrow0 = qb * BQ + wave * 16;

    // ---- Q in A layout; scale folded with log2(e) ----
    const float scale = 0.088388347648318447f * 1.44269504088896340736f;
    v16h qa[4];
#pragma unroll
    for (int c = 0; c < 4; ++c) {
        const float* qp = Qp + (size_t)(qrow0 + l16) * ATT_D + c * 32 + h16 * 8;
        v4f q0 = *(const v4f*)(qp);
        v4f q1 = *(const v4f*)(qp + 4);
        v4f q2 = *(const v4f*)(qp + 16);
        v4f q3 = *(const v4f*)(qp + 20);
        v16h a;
#pragma unroll
        for (int j = 0; j < 4; ++j) {
            a[j]      = (_Float16)(q0[j] * scale);
            a[4 + j]  = (_Float16)(q1[j] * scale);
            a[8 + j]  = (_Float16)(q2[j] * scale);
            a[12 + j] = (_Float16)(q3[j] * scale);
        }
        qa[c] = a;
    }

    v8f acc[8];
#pragma unroll
    for (int f = 0; f < 8; ++f) acc[f] = (v8f)0.0f;
    v8f accL = (v8f)0.0f;            // running row-sums (ones-trick)
    float mrun[8];
#pragma unroll
    for (int v = 0; v < 8; ++v) mrun[v] = -1e30f;

    // ---- staging: each thread owns 4 key rows x 8 d-columns ----
    const int r0 = (tid >> 4) * 4;    // key rows r0..r0+3
    const int c0 = (tid & 15) * 8;    // d columns c0..c0+7
    const float* Kst = Kp + (size_t)r0 * ATT_D + c0;
    const float* Vst = Vp + (size_t)r0 * ATT_D + c0;

    v4f kreg[8], vreg[8];             // [row*2 + half]
    auto load_tile = [&](int kb) {
        const float* kp = Kst + (size_t)kb * ATT_D;
        const float* vp = Vst + (size_t)kb * ATT_D;
#pragma unroll
        for (int row = 0; row < 4; ++row) {
            kreg[row * 2 + 0] = *(const v4f*)(kp + row * ATT_D);
            kreg[row * 2 + 1] = *(const v4f*)(kp + row * ATT_D + 4);
            vreg[row * 2 + 0] = *(const v4f*)(vp + row * ATT_D);
            vreg[row * 2 + 1] = *(const v4f*)(vp + row * ATT_D + 4);
        }
    };
    auto store_tile = [&]() {
        // K row-major: pack adjacent d pairs -> ds_store_b32
#pragma unroll
        for (int row = 0; row < 4; ++row) {
#pragma unroll
            for (int i = 0; i < 4; ++i) {
                v2h pk;
                pk[0] = (_Float16)kreg[row * 2 + (i >> 1)][(i & 1) * 2 + 0];
                pk[1] = (_Float16)kreg[row * 2 + (i >> 1)][(i & 1) * 2 + 1];
                *(v2h*)&sK[(r0 + row) * ATT_D + c0 + i * 2] = pk;
            }
        }
        // V transposed: pack adjacent key pairs -> ds_store_b32
#pragma unroll
        for (int i = 0; i < 8; ++i) {
#pragma unroll
            for (int p2 = 0; p2 < 2; ++p2) {
                v2h pk;
                pk[0] = (_Float16)vreg[(p2 * 2 + 0) * 2 + (i >> 2)][i & 3];
                pk[1] = (_Float16)vreg[(p2 * 2 + 1) * 2 + (i >> 2)][i & 3];
                *(v2h*)&sVt[(c0 + i) * BK + r0 + 2 * p2] = pk;
            }
        }
    };

    const int ntiles = 2 * (qb + 1);   // causal bound for this q block
    load_tile(0);
    store_tile();
    __syncthreads();

    _Float16* pp = &sP[wave * 16 * BK];
    for (int t = 0; t < ntiles; ++t) {
        const int kb = t * BK;
        const bool more = (t + 1 < ntiles);
        if (more) load_tile(kb + BK);     // HBM loads in flight under the WMMAs

        if (kb <= qrow0 + 15) {           // wave-level skip of dead tiles
            if (kb + BK - 1 > qrow0)      // only diagonal tiles pay masking
                attn_tile<true >(sK, sVt, pp, qa, acc, accL, mrun, kb, qrow0, l16, h16);
            else
                attn_tile<false>(sK, sVt, pp, qa, acc, accL, mrun, kb, qrow0, l16, h16);
        }
        __syncthreads();                  // all reads of sK/sVt complete
        if (more) {
            store_tile();                 // convert+stage next tile
            __syncthreads();              // stores visible before next compute
        }
    }

    // ---- epilogue: normalize by accL (row sums) and store ----
#pragma unroll
    for (int v = 0; v < 8; ++v) {
        const float inv = 1.0f / accL[v];
        const int qm = qrow0 + v + 8 * h16;
        float* orow = Op + (size_t)qm * ATT_D;
#pragma unroll
        for (int f = 0; f < 8; ++f)
            orow[f * 16 + l16] = acc[f][v] * inv;
    }
}

extern "C" void kernel_launch(void* const* d_in, const int* in_sizes, int n_in,
                              void* d_out, int out_size, void* d_ws, size_t ws_size,
                              hipStream_t stream) {
    (void)in_sizes; (void)n_in; (void)out_size; (void)d_ws; (void)ws_size;
    const float* q = (const float*)d_in[0];
    const float* k = (const float*)d_in[1];
    const float* v = (const float*)d_in[2];
    // d_in[3] is the causal mask; causality is applied analytically in-kernel.
    float* o = (float*)d_out;
    dim3 grid(ATT_S / BQ, ATT_B * ATT_H);
    fa2_wmma_kernel<<<grid, dim3(256), 0, stream>>>(q, k, v, o);
}
